// MaskedGCN_15264313770213
// MI455X (gfx1250) — matrix operand order repaired and checked by reading, not compiled, not run on hardware
//
#include <hip/hip_runtime.h>
#include <hip/hip_bf16.h>

// ---------------------------------------------------------------------------
// MaskedGCN on MI455X (gfx1250, wave32):
//   h   = x @ bf16(W1*M1)              -> v_wmma_f32_16x16x32_bf16 (async-LDS A)
//   agg = scatter(h[src]*norm -> dst)  -> global f32 atomics (L2 resident)
//   r   = relu(agg + selfloop + b1)
//   out = r @ bf16(pad(W2*M2)) + b2    -> v_wmma_f32_16x16x32_bf16
// B matrices are pre-swizzled into per-lane 32B WMMA fragments so operand
// loads are plain global_load_b128 (no v_mov_b16 packing).
// ---------------------------------------------------------------------------

typedef __attribute__((ext_vector_type(16))) __bf16 v16bf;
typedef __attribute__((ext_vector_type(8)))  float  v8f;

#define N_NODES 100000
#define N_EDGES 3200000
#define N_FEAT  512
#define N_DIMS  64
#define N_CLSS  10
#define N_TILES (N_NODES / 16)   // 6250 exactly

// ---- workspace layout (bytes, 32B-aligned where needed) -------------------
#define WS_DEG   0                         // 100000 f32 (deg, then dinv in place)
#define WS_H     400384                    // 100000*64 f32
#define WS_AGG   26000384                  // 100000*64 f32
#define WS_B1    51600384                  // swizzled bf16 B1: [16][4][32][16]
#define WS_W2    51665920                  // swizzled bf16 W2: [2][32][16]

// ---- CDNA5 async global->LDS path (guarded; sync fallback compiles) -------
#if __has_builtin(__builtin_amdgcn_global_load_async_to_lds_b128) && \
    __has_builtin(__builtin_amdgcn_s_wait_asynccnt)
#define HAVE_ASYNC_LDS 1
// Exact types from clang's diagnostic: param0 is 16B int vector in AS(1)
// (printed as "__device__"), param1 the LDS-side mirror in AS(3).
typedef int v4i __attribute__((vector_size(4 * sizeof(int))));
typedef __attribute__((address_space(1))) v4i* gv4i_p;
typedef __attribute__((address_space(3))) v4i* lv4i_p;
#else
#define HAVE_ASYNC_LDS 0
#endif

// ---------------------------------------------------------------------------
__global__ void zero_f32(float* __restrict__ p, long n) {
  long i = (long)blockIdx.x * blockDim.x + threadIdx.x;
  long stride = (long)gridDim.x * blockDim.x;
  for (; i < n; i += stride) p[i] = 0.0f;
}

// Fold masks into weights, convert to bf16, and SWIZZLE into WMMA B-fragment
// order: element b[2v+j] of fragment (kstep, nt, lane) is B[k][n] with
//   k = kstep*32 + 2*(v&3) + 16*(v>>2) + 8*(lane>=16) + j,  n = nt*16 + (lane&15)
// stored contiguously 16 bf16 (32B) per (kstep, nt, lane).
__global__ void prep_weights(const float* __restrict__ W1, const float* __restrict__ M1,
                             const float* __restrict__ W2, const float* __restrict__ M2,
                             __bf16* __restrict__ Bsw, __bf16* __restrict__ Wsw) {
  int i = blockIdx.x * blockDim.x + threadIdx.x;
  if (i < 16 * 4 * 32 * 16) {                 // gemm1 B: [kstep][nt][lane][16]
    const int idx  = i & 15;
    const int lane = (i >> 4) & 31;
    const int nt   = (i >> 9) & 3;
    const int kst  = i >> 11;
    const int v = idx >> 1, j = idx & 1;
    const int k = kst * 32 + 2 * (v & 3) + 16 * (v >> 2) + 8 * (lane >> 4) + j;
    const int n = nt * 16 + (lane & 15);
    Bsw[i] = (__bf16)(W1[k * N_DIMS + n] * M1[k * N_DIMS + n]);
  }
  if (i < 2 * 32 * 16) {                      // classifier B: [kstep][lane][16]
    const int idx  = i & 15;
    const int lane = (i >> 4) & 31;
    const int kst  = i >> 9;
    const int v = idx >> 1, j = idx & 1;
    const int k = kst * 32 + 2 * (v & 3) + 16 * (v >> 2) + 8 * (lane >> 4) + j;
    const int n = lane & 15;
    const float val = (n < N_CLSS) ? W2[k * N_CLSS + n] * M2[k * N_CLSS + n] : 0.0f;
    Wsw[i] = (__bf16)val;
  }
}

// deg[dst] += w  (3.2M scalar f32 atomics into 400KB -> L2 resident)
__global__ void degree_kernel(const long long* __restrict__ ei,
                              const float* __restrict__ ew,
                              float* __restrict__ deg) {
  int e = blockIdx.x * blockDim.x + threadIdx.x;
  if (e >= N_EDGES) return;
  atomicAdd(&deg[(int)ei[N_EDGES + e]], ew[e]);
}

// dinv = deg>0 ? rsqrt(max(deg,1e-12)) : 0   (self-loop weight 1 added here)
__global__ void dinv_kernel(float* __restrict__ deg) {
  int i = blockIdx.x * blockDim.x + threadIdx.x;
  if (i >= N_NODES) return;
  float d = deg[i] + 1.0f;
  deg[i] = (d > 0.0f) ? rsqrtf(fmaxf(d, 1e-12f)) : 0.0f;
}

// ---------------------------------------------------------------------------
// h[100000,64] = x[100000,512] @ B1[512,64]  via bf16 WMMA.
// One wave per 16-row M-tile, 4 N-tiles, K stepped by 32.
// A tile: f32 staged in LDS (double-buffered async global->LDS b128 when the
// toolchain exposes the builtin, coalesced sync loads otherwise), converted
// to bf16 at fragment-build time. B: pre-swizzled 32B fragments (2x b128).
// No block barriers -> early-exit waves are safe.
// ---------------------------------------------------------------------------
__global__ void __launch_bounds__(256)
gemm1_wmma(const float* __restrict__ x, const __bf16* __restrict__ Bsw,
           float* __restrict__ h) {
  __shared__ float Lx[8][2][16][36];   // per-wave double-buffered 16x32 f32 tile
                                       // row stride 144B: 16B/8B aligned accesses
  const int lane = threadIdx.x & 31;
  const int wv   = threadIdx.x >> 5;
  const int tile = blockIdx.x * 8 + wv;
  if (tile >= N_TILES) return;               // wave-uniform exit
  const int row0 = tile * 16;
  const int half = lane >> 4;                // 0: lanes 0-15, 1: lanes 16-31
  const int l15  = lane & 15;

  v8f c[4];
  #pragma unroll
  for (int nt = 0; nt < 4; ++nt) c[nt] = (v8f){};

#if HAVE_ASYNC_LDS
  // each group: 4 x async b128 -> 2KB tile (lane -> row = i*4 + lane/8, 4 floats)
  const int arow = lane >> 3;
  const int acol = (lane & 7) * 4;
  auto issue_async = [&](int buf, int ks) {
    #pragma unroll
    for (int i = 0; i < 4; ++i) {
      const int row = i * 4 + arow;
      __builtin_amdgcn_global_load_async_to_lds_b128(
          (gv4i_p)(x + (long)(row0 + row) * N_FEAT + ks + acol),
          (lv4i_p)&Lx[wv][buf][row][acol], 0, 0);
    }
  };
  issue_async(0, 0);
  int buf = 0;
#endif

  for (int ks = 0; ks < N_FEAT; ks += 32) {
#if HAVE_ASYNC_LDS
    if (ks + 32 < N_FEAT) {
      issue_async(buf ^ 1, ks + 32);              // prefetch next tile
      __builtin_amdgcn_s_wait_asynccnt(4);        // wait for current tile only
    } else {
      __builtin_amdgcn_s_wait_asynccnt(0);
    }
    asm volatile("" ::: "memory");
    const int rb = buf;
#else
    #pragma unroll
    for (int r = 0; r < 16; ++r)                  // coalesced 128B per row
      Lx[wv][0][r][lane] = x[(long)(row0 + r) * N_FEAT + ks + lane];
    if (ks + 32 < N_FEAT)
      __builtin_prefetch(&x[(long)(row0 + l15) * N_FEAT + ks + 32], 0, 3);
    const int rb = 0;
#endif

    // ---- A fragment: ISA 16-bit A 16x32 layout ----------------------------
    // lanes 0-15:  V0..3 -> K 0..7,  V4..7 -> K 16..23
    // lanes 16-31: V0..3 -> K 8..15, V4..7 -> K 24..31
    v16bf a;
    #pragma unroll
    for (int v = 0; v < 8; ++v) {
      const int k0 = 2 * (v & 3) + ((v >= 4) ? 16 : 0) + 8 * half;
      const float2 p = *(const float2*)&Lx[wv][rb][l15][k0];
      a[2 * v + 0] = (__bf16)p.x;
      a[2 * v + 1] = (__bf16)p.y;
    }

    // ---- B fragments: pre-swizzled, one 32B vector load each --------------
    const __bf16* Bk = Bsw + (long)(ks >> 5) * (4 * 32 * 16);
    #pragma unroll
    for (int nt = 0; nt < 4; ++nt) {
      const v16bf b = *(const v16bf*)(Bk + (nt * 32 + lane) * 16);
      c[nt] = __builtin_amdgcn_wmma_f32_16x16x32_bf16(
          false, a, false, b, (short)0, c[nt], false, false);
    }
#if HAVE_ASYNC_LDS
    buf ^= 1;
#endif
  }

  // ---- store D: lanes 0-15 N=lane M=r, lanes 16-31 N=lane-16 M=r+8 --------
  #pragma unroll
  for (int nt = 0; nt < 4; ++nt)
    #pragma unroll
    for (int r = 0; r < 8; ++r)
      h[(long)(row0 + r + 8 * half) * N_DIMS + nt * 16 + l15] = c[nt][r];
}

// ---------------------------------------------------------------------------
// agg[dst] += h[src] * (dinv[src]*w*dinv[dst])  -- one wave per edge,
// each lane owns a float2 of the 64-feature row -> 2 f32 atomics per lane.
// ---------------------------------------------------------------------------
__global__ void scatter_kernel(const long long* __restrict__ ei,
                               const float* __restrict__ ew,
                               const float* __restrict__ dinv,
                               const float* __restrict__ h,
                               float* __restrict__ agg) {
  const int e    = (int)((blockIdx.x * blockDim.x + threadIdx.x) >> 5);
  const int lane = threadIdx.x & 31;
  if (e >= N_EDGES) return;
  const int src = (int)ei[e];
  const int dst = (int)ei[N_EDGES + e];
  const float norm = dinv[src] * ew[e] * dinv[dst];
  const float2 hv = *(const float2*)(h + (long)src * N_DIMS + lane * 2);
  float* ap = agg + (long)dst * N_DIMS + lane * 2;
  atomicAdd(ap + 0, hv.x * norm);
  atomicAdd(ap + 1, hv.y * norm);
}

// r = relu(agg + h*dinv^2 (self-loop) + b1), in place in agg
__global__ void relu_kernel(const float* __restrict__ h,
                            const float* __restrict__ dinv,
                            const float* __restrict__ b1,
                            float* __restrict__ agg) {
  long i = (long)blockIdx.x * blockDim.x + threadIdx.x;
  if (i >= (long)N_NODES * N_DIMS) return;
  const long node = i >> 6;
  const int  f    = (int)(i & 63);
  const float di = dinv[node];
  agg[i] = fmaxf(agg[i] + h[i] * di * di + b1[f], 0.0f);
}

// ---------------------------------------------------------------------------
// out[100000,10] = r[100000,64] @ W2p[64,16(pad)] + b2 via 2 bf16 WMMAs/wave.
// ---------------------------------------------------------------------------
__global__ void __launch_bounds__(256)
classifier_wmma(const float* __restrict__ r, const __bf16* __restrict__ Wsw,
                const float* __restrict__ b2, float* __restrict__ out) {
  const int lane = threadIdx.x & 31;
  const int wv   = threadIdx.x >> 5;
  const int tile = blockIdx.x * 8 + wv;
  if (tile >= N_TILES) return;
  const int row0 = tile * 16;
  const int half = lane >> 4;
  const int l15  = lane & 15;

  v8f c = {};
  #pragma unroll
  for (int ks = 0; ks < N_DIMS; ks += 32) {
    v16bf a;
    #pragma unroll
    for (int v = 0; v < 8; ++v) {
      const int ka = ks + 2 * (v & 3) + ((v >= 4) ? 16 : 0) + 8 * half;
      const float2 p = *(const float2*)(r + (long)(row0 + l15) * N_DIMS + ka);
      a[2 * v + 0] = (__bf16)p.x;
      a[2 * v + 1] = (__bf16)p.y;
    }
    const v16bf b = *(const v16bf*)(Wsw + ((ks >> 5) * 32 + lane) * 16);
    c = __builtin_amdgcn_wmma_f32_16x16x32_bf16(
        false, a, false, b, (short)0, c, false, false);
  }

  if (l15 < N_CLSS) {                 // divergence only after the WMMAs
    const float bias = b2[l15];
    #pragma unroll
    for (int rr = 0; rr < 8; ++rr)
      out[(row0 + rr + 8 * half) * N_CLSS + l15] = c[rr] + bias;
  }
}

// ---------------------------------------------------------------------------
extern "C" void kernel_launch(void* const* d_in, const int* in_sizes, int n_in,
                              void* d_out, int out_size, void* d_ws, size_t ws_size,
                              hipStream_t stream) {
  const float*     x  = (const float*)d_in[0];
  const long long* ei = (const long long*)d_in[1];   // int64 edge_index [2, E]
  const float*     ew = (const float*)d_in[2];
  const float*     W1 = (const float*)d_in[3];
  const float*     M1 = (const float*)d_in[4];
  const float*     b1 = (const float*)d_in[5];
  const float*     W2 = (const float*)d_in[6];
  const float*     M2 = (const float*)d_in[7];
  const float*     b2 = (const float*)d_in[8];
  float* out = (float*)d_out;

  char* ws = (char*)d_ws;
  float*  deg = (float*)(ws + WS_DEG);   // becomes dinv in place
  float*  h   = (float*)(ws + WS_H);
  float*  agg = (float*)(ws + WS_AGG);
  __bf16* Bsw = (__bf16*)(ws + WS_B1);
  __bf16* Wsw = (__bf16*)(ws + WS_W2);

  zero_f32<<<2048, 256, 0, stream>>>(deg, (long)N_NODES);
  zero_f32<<<8192, 256, 0, stream>>>(agg, (long)N_NODES * N_DIMS);
  prep_weights<<<(16 * 4 * 32 * 16 + 255) / 256, 256, 0, stream>>>(W1, M1, W2, M2, Bsw, Wsw);
  degree_kernel<<<(N_EDGES + 255) / 256, 256, 0, stream>>>(ei, ew, deg);
  dinv_kernel<<<(N_NODES + 255) / 256, 256, 0, stream>>>(deg);
  gemm1_wmma<<<(N_TILES + 7) / 8, 256, 0, stream>>>(x, Bsw, h);
  scatter_kernel<<<N_EDGES / 8, 256, 0, stream>>>(ei, ew, deg, h, agg);
  relu_kernel<<<(int)(((long)N_NODES * N_DIMS + 255) / 256), 256, 0, stream>>>(h, deg, b1, agg);
  classifier_wmma<<<(N_TILES + 7) / 8, 256, 0, stream>>>(agg, Wsw, b2, out);
}
